// SLAYFeatures_51986284151002
// MI455X (gfx1250) — matrix operand order repaired and checked by reading, not compiled, not run on hardware
//
#include <hip/hip_runtime.h>
#include <hip/hip_bf16.h>

typedef __attribute__((ext_vector_type(16))) _Float16 v16h;
typedef __attribute__((ext_vector_type(8)))  float    v8f;

#define WMMA_F16(a, b, c) \
  __builtin_amdgcn_wmma_f32_16x16x32_f16(false, (a), false, (b), (short)0, (c), false, false)

// Problem constants (match reference)
constexpr int HEADS = 16;
constexpr int HD    = 64;    // head dim (K)
constexpr int NF    = 32;    // random features (M)
constexpr int PD    = 16;    // poly dim (P)
constexpr int RQ    = 2;     // quadrature nodes
constexpr int ROW_F = RQ * HEADS * PD * NF;  // 16384 floats per output row

// LDS strides (floats). Chosen so lanes0-15 vs 16-31 hit disjoint bank sets
// and float4 reads stay 16B-aligned (stride % 4 == 0 for prf).
constexpr int PSTR   = 20;                        // poly row stride
constexpr int FSTR   = 36;                        // prf row stride
constexpr int WAVE_F = PD * PSTR + RQ * 16 * FSTR; // 320 + 1152 = 1472 floats/wave

__global__ __launch_bounds__(256)
void slay_feat_kernel(const float* __restrict__ x,
                      const float* __restrict__ omega,
                      const float* __restrict__ anchors,
                      const float* __restrict__ qn,
                      const float* __restrict__ qw,
                      float* __restrict__ out)
{
  __shared__ float inv_s[16 * 16];          // [row][head] inverse norms
  __shared__ float feat[8 * WAVE_F];        // per-wave feature staging

  const int tid  = threadIdx.x;
  const int wave = tid >> 5;
  const int lane = tid & 31;
  const int b0   = blockIdx.x * 16;         // 16 batch rows per block

  // ---------- phase 1: per-(row, head) inverse L2 norms ----------
  {
    const int row = tid >> 4, hh = tid & 15;
    const float4* p = (const float4*)(x + (size_t)(b0 + row) * 1024 + hh * HD);
    float s = 0.f;
#pragma unroll
    for (int i = 0; i < 16; ++i) {
      float4 v = p[i];
      s += v.x * v.x + v.y * v.y + v.z * v.z + v.w * v.w;
    }
    inv_s[row * 16 + hh] = 1.0f / fmaxf(sqrtf(s), 1e-4f);
  }
  __syncthreads();

  // quadrature parameters
  float sN[2], sq2s[2], scl[2];
#pragma unroll
  for (int r = 0; r < RQ; ++r) {
    sN[r]   = fmaxf(qn[r], 1e-6f);
    sq2s[r] = sqrtf(2.0f * sN[r]);
    scl[r]  = sqrtf(fmaxf(qw[r], 1e-6f)) * rsqrtf(32.0f + 1e-6f);
  }

  const int mrow = lane & 15;   // matrix row (A) / column (B,C,D)
  const int hi   = lane >> 4;   // lane half selects K sub-block

  float* polyS = feat + wave * WAVE_F;   // [16 rows][PSTR]
  float* prfS  = polyS + PD * PSTR;      // [r][16 rows][FSTR]

  for (int hh = 0; hh < 2; ++hh) {
    const int h = wave + hh * 8;

    // ---- A fragments: xn batch tile, 16x64 as two K=32 f16 fragments ----
    // A 16x32 in contiguous k order (k=16h+s): lanes0-15 row M=lane, v_i holds K=2i,2i+1;
    // lanes16-31 same rows, K offset +16.
    const float  invn = inv_s[mrow * 16 + h];
    const float* xr   = x + (size_t)(b0 + mrow) * 1024 + h * HD;
    v16h A0, A1;
#pragma unroll
    for (int i = 0; i < 8; ++i) {
      int k = 2 * i + hi * 16;
      float2 v = *(const float2*)(xr + k);
      A0[2 * i]     = (_Float16)(v.x * invn);
      A0[2 * i + 1] = (_Float16)(v.y * invn);
      float2 w = *(const float2*)(xr + k + 32);
      A1[2 * i]     = (_Float16)(w.x * invn);
      A1[2 * i + 1] = (_Float16)(w.y * invn);
    }

    // ---- B fragments: anchors^T (K=d, N=p), 64x16 as two 32x16 tiles ----
    // B 32x16, same contiguous k order as A: lane n = lane&15, v_i holds K=2i,2i+1 (+16 for hi lanes)
    const float* an = anchors + mrow * HD;
    v16h Ba0, Ba1;
#pragma unroll
    for (int i = 0; i < 8; ++i) {
      int k = 2 * i + hi * 16;
      float2 v = *(const float2*)(an + k);
      Ba0[2 * i]     = (_Float16)v.x;
      Ba0[2 * i + 1] = (_Float16)v.y;
      float2 w = *(const float2*)(an + k + 32);
      Ba1[2 * i]     = (_Float16)w.x;
      Ba1[2 * i + 1] = (_Float16)w.y;
    }

    // poly_proj = xn . anchors^T  (accumulate over the two K=32 chunks)
    v8f cp = {};
    cp = WMMA_F16(A0, Ba0, cp);
    cp = WMMA_F16(A1, Ba1, cp);

    // poly_feat = clip(proj,-1,1)^2 / sqrt(16)
#pragma unroll
    for (int i = 0; i < 8; ++i) {
      float v = fminf(fmaxf(cp[i], -1.0f), 1.0f);
      polyS[(i + hi * 8) * PSTR + mrow] = v * v * 0.25f;
    }

    // ---- prf features: xn . omega[r,h] (64x32), two N=16 tiles per r ----
#pragma unroll
    for (int r = 0; r < RQ; ++r) {
#pragma unroll
      for (int nb = 0; nb < 2; ++nb) {
        const float* om = omega + (size_t)(r * HEADS + h) * HD * NF + nb * 16 + mrow;
        v16h Bo0, Bo1;
#pragma unroll
        for (int i = 0; i < 8; ++i) {
          int k = 2 * i + hi * 16;
          Bo0[2 * i]     = (_Float16)om[(size_t)k * NF];
          Bo0[2 * i + 1] = (_Float16)om[(size_t)(k + 1) * NF];
          Bo1[2 * i]     = (_Float16)om[(size_t)(k + 32) * NF];
          Bo1[2 * i + 1] = (_Float16)om[(size_t)(k + 33) * NF];
        }
        v8f cf = {};
        cf = WMMA_F16(A0, Bo0, cf);
        cf = WMMA_F16(A1, Bo1, cf);

        // prf_feat = exp(clip(proj*sqrt(2s) - s, +-20)) * sqrt(w)/sqrt(32+eps)
#pragma unroll
        for (int i = 0; i < 8; ++i) {
          float a = cf[i] * sq2s[r] - sN[r];
          a = fminf(fmaxf(a, -20.0f), 20.0f);
          prfS[r * (16 * FSTR) + (i + hi * 8) * FSTR + nb * 16 + mrow] =
              __expf(a) * scl[r];
        }
      }
    }

    __syncthreads();

    // ---- fused outer product + coalesced b128 stores ----
    // out index per row: r*8192 + h*512 + p*32 + m
#pragma unroll 1
    for (int row = 0; row < 16; ++row) {
#pragma unroll
      for (int r = 0; r < RQ; ++r) {
        float*       po = out + (size_t)(b0 + row) * ROW_F + r * 8192 + h * 512;
        const float* pr = prfS + r * (16 * FSTR) + row * FSTR;
        const float* pp = polyS + row * PSTR;
#pragma unroll
        for (int it = 0; it < 4; ++it) {
          int pm = it * 128 + lane * 4;            // 0..511, float4 granules
          float  pl = pp[pm >> 5];                 // poly[p]
          float4 pf = *(const float4*)(pr + (pm & 31));  // prf[m..m+3]
          float4 o  = make_float4(pl * pf.x, pl * pf.y, pl * pf.z, pl * pf.w);
          *(float4*)(po + pm) = o;
        }
      }
    }

    __syncthreads();
  }
}

extern "C" void kernel_launch(void* const* d_in, const int* in_sizes, int n_in,
                              void* d_out, int out_size, void* d_ws, size_t ws_size,
                              hipStream_t stream) {
  const float* x       = (const float*)d_in[0];
  const float* omega   = (const float*)d_in[1];
  const float* anchors = (const float*)d_in[2];
  const float* qn      = (const float*)d_in[3];
  const float* qw      = (const float*)d_in[4];
  float* out = (float*)d_out;

  const int B = in_sizes[0] / 1024;   // 4096
  slay_feat_kernel<<<B / 16, 256, 0, stream>>>(x, omega, anchors, qn, qw, out);
}
